// VectorQuantizer_2774548873906
// MI455X (gfx1250) — matrix operand order repaired and checked by reading, not compiled, not run on hardware
//
#include <hip/hip_runtime.h>
#include <hip/hip_bf16.h>

typedef __attribute__((ext_vector_type(16))) _Float16 v16h;
typedef __attribute__((ext_vector_type(8)))  _Float16 v8h;
typedef __attribute__((ext_vector_type(8)))  float    v8f;

#define VQ_D   512
#define VQ_K   8192
#define VQ_N   16384
#define VQ_ND  8388608LL          // N * D
#define VQ_KT  (VQ_K / 16)        // 512 code tiles

// ---- workspace layout (bytes) ----
#define WS_A16    0                           // N*D halves  = 16,777,216 B
#define WS_B16    16777216                    // K*D halves  =  8,388,608 B
#define WS_E2     25165824                    // K floats    =     32,768 B
#define WS_IDX    25198592                    // N ints      =     65,536 B
#define WS_ACC    25264128                    // 1 float

// ---------------------------------------------------------------------------
// 1) inputs f32 -> f16 (8 elems/thread), zero the loss accumulator
// ---------------------------------------------------------------------------
__global__ void vq_convert_inputs(const float* __restrict__ in,
                                  _Float16* __restrict__ a16,
                                  float* __restrict__ accum) {
    if (blockIdx.x == 0 && threadIdx.x == 0) *accum = 0.0f;
    long long t = (long long)blockIdx.x * blockDim.x + threadIdx.x;  // ND/8 threads
    const float4* p = (const float4*)in + t * 2;
    float4 a = p[0], b = p[1];
    v8h h;
    h[0] = (_Float16)a.x; h[1] = (_Float16)a.y;
    h[2] = (_Float16)a.z; h[3] = (_Float16)a.w;
    h[4] = (_Float16)b.x; h[5] = (_Float16)b.y;
    h[6] = (_Float16)b.z; h[7] = (_Float16)b.w;
    *((v8h*)a16 + t) = h;
}

// ---------------------------------------------------------------------------
// 2) codebook f32 -> f16 + per-code squared norm (one wave32 per code row)
// ---------------------------------------------------------------------------
__global__ void vq_convert_codebook(const float* __restrict__ cb,
                                    _Float16* __restrict__ b16,
                                    float* __restrict__ e2) {
    const int lane = threadIdx.x & 31;
    const int wave = threadIdx.x >> 5;
    const int row  = blockIdx.x * 8 + wave;          // 1024 blocks * 8 waves = 8192 rows
    const float* src = cb + (long long)row * VQ_D + lane * 16;
    float s = 0.0f;
    v8h h0, h1;
#pragma unroll
    for (int i = 0; i < 8; ++i) { float v = src[i];     s += v * v; h0[i] = (_Float16)v; }
#pragma unroll
    for (int i = 0; i < 8; ++i) { float v = src[8 + i]; s += v * v; h1[i] = (_Float16)v; }
    _Float16* dst = b16 + (long long)row * VQ_D + lane * 16;
    *(v8h*)dst       = h0;
    *(v8h*)(dst + 8) = h1;
#pragma unroll
    for (int m = 16; m >= 1; m >>= 1) s += __shfl_xor(s, m, 32);
    if (lane == 0) e2[row] = s;
}

// ---------------------------------------------------------------------------
// 3) WMMA distance + argmin.  256 threads = 8 wave32, 128 rows of x per block.
//    Each wave keeps its full 16x512 f16 A tile register-resident (16 frags =
//    128 VGPRs; launch_bounds(256,1) lifts the VGPR cap so nothing spills) and
//    streams all 8192 codes as B tiles through L0/L2, two interleaved WMMA
//    accumulation chains along K.
// ---------------------------------------------------------------------------
__global__ void __launch_bounds__(256, 1)
vq_argmin_wmma(const _Float16* __restrict__ a16,
               const _Float16* __restrict__ b16,
               const float* __restrict__ e2,
               int* __restrict__ idxw,
               float* __restrict__ out) {
    extern __shared__ _Float16 smem[];               // 128 rows * 512 halves = 128 KB
    const int lane = threadIdx.x & 31;
    const int wave = threadIdx.x >> 5;
    const int m    = lane & 15;                      // A row / B col within tile
    const int kbA  = (lane >> 4) << 3;               // A frag: K base 0 or 8 (then +16)
    const int kbB  = (lane >> 4) << 3;               // B frag: K base 0 or 8 (then +16)

    // cooperative stage of this block's 128 input rows into LDS
    {
        const uint4* g = (const uint4*)(a16 + (long long)blockIdx.x * 128 * VQ_D);
        uint4* l = (uint4*)smem;
        for (int i = threadIdx.x; i < 128 * VQ_D / 8; i += 256) l[i] = g[i];
    }
    __syncthreads();

    // load the wave's entire A tile into registers (ISA A-fragment layout)
    const _Float16* aRow = smem + (wave * 16 + m) * VQ_D;
    v16h afrag[16];
#pragma unroll
    for (int kc = 0; kc < 16; ++kc) {
        const _Float16* ap = aRow + kc * 32 + kbA;
        v8h a0 = *(const v8h*)ap;
        v8h a1 = *(const v8h*)(ap + 16);
        afrag[kc] = __builtin_shufflevector(a0, a1,
                        0,1,2,3,4,5,6,7,8,9,10,11,12,13,14,15);
    }

    float best[8];
    int   bidx[8];
#pragma unroll
    for (int r = 0; r < 8; ++r) { best[r] = 3.4e38f; bidx[r] = 0; }

    for (int j = 0; j < VQ_KT; ++j) {
        if (j + 1 < VQ_KT)
            __builtin_prefetch(b16 + (long long)(j + 1) * 16 * VQ_D, 0, 1);

        const _Float16* bRow = b16 + (long long)(j * 16 + m) * VQ_D;
        v8f acc0 = {};
        v8f acc1 = {};
#pragma unroll
        for (int kc = 0; kc < 16; kc += 2) {
            const _Float16* bp0 = bRow + kc * 32 + kbB;
            v8h b00 = *(const v8h*)bp0;
            v8h b01 = *(const v8h*)(bp0 + 16);
            const _Float16* bp1 = bRow + (kc + 1) * 32 + kbB;
            v8h b10 = *(const v8h*)bp1;
            v8h b11 = *(const v8h*)(bp1 + 16);
            v16h bv0 = __builtin_shufflevector(b00, b01,
                           0,1,2,3,4,5,6,7,8,9,10,11,12,13,14,15);
            v16h bv1 = __builtin_shufflevector(b10, b11,
                           0,1,2,3,4,5,6,7,8,9,10,11,12,13,14,15);
            acc0 = __builtin_amdgcn_wmma_f32_16x16x32_f16(
                       false, afrag[kc],     false, bv0, (short)0, acc0, false, false);
            acc1 = __builtin_amdgcn_wmma_f32_16x16x32_f16(
                       false, afrag[kc + 1], false, bv1, (short)0, acc1, false, false);
        }
        // dist (up to a per-row constant) = ||e||^2 - 2 * x.e
        const int   code = j * 16 + m;
        const float e2c  = e2[code];
#pragma unroll
        for (int r = 0; r < 8; ++r) {
            float dv = fmaf(-2.0f, acc0[r] + acc1[r], e2c);
            if (dv < best[r]) { best[r] = dv; bidx[r] = code; }
        }
    }

    // reduce across the 16 lanes of each half (codes live across lanes);
    // tie-break on smallest index to match jnp.argmin semantics
#pragma unroll
    for (int r = 0; r < 8; ++r) {
        float bv = best[r];
        int   bi = bidx[r];
#pragma unroll
        for (int mk = 1; mk < 16; mk <<= 1) {
            float ov = __shfl_xor(bv, mk, 32);
            int   oi = __shfl_xor(bi, mk, 32);
            if (ov < bv || (ov == bv && oi < bi)) { bv = ov; bi = oi; }
        }
        if ((lane & 15) == 0) {
            int rowg = blockIdx.x * 128 + wave * 16 + ((lane >> 4) << 3) + r;
            idxw[rowg] = bi;
            out[2 * VQ_ND + 2 + rowg] = (float)bi;   // idx output (as f32)
        }
    }
}

// ---------------------------------------------------------------------------
// 4) gather codebook rows, write quantized_out + quantized, accumulate MSE sum
// ---------------------------------------------------------------------------
__global__ void vq_gather_loss(const float* __restrict__ in,
                               const float* __restrict__ cb,
                               const int* __restrict__ idxw,
                               float* __restrict__ out,
                               float* __restrict__ accum) {
    long long t    = (long long)blockIdx.x * blockDim.x + threadIdx.x;  // ND/4 threads
    long long elem = t * 4;
    int row = (int)(elem >> 9);
    int col = (int)(elem & 511);
    int ci  = idxw[row];

    float4 q = *(const float4*)(cb + (long long)ci * VQ_D + col);
    float4 x = *(const float4*)(in + elem);

    *(float4*)(out + elem) = q;                       // quantized_out == quantized
    float* qz = out + VQ_ND + 2 + elem;               // quantized (8-byte aligned base)
    float2 q01; q01.x = q.x; q01.y = q.y;
    float2 q23; q23.x = q.z; q23.y = q.w;
    *(float2*)qz       = q01;
    *(float2*)(qz + 2) = q23;

    float dx = q.x - x.x, dy = q.y - x.y, dz = q.z - x.z, dw = q.w - x.w;
    float s = dx * dx + dy * dy + dz * dz + dw * dw;
#pragma unroll
    for (int m = 16; m >= 1; m >>= 1) s += __shfl_xor(s, m, 32);
    if ((threadIdx.x & 31) == 0) atomicAdd(accum, s);
}

// ---------------------------------------------------------------------------
// 5) finalize both (numerically identical) losses
// ---------------------------------------------------------------------------
__global__ void vq_finalize(const float* __restrict__ accum, float* __restrict__ out) {
    if (threadIdx.x == 0) {
        float loss = *accum * (1.0f / (float)VQ_ND);
        out[VQ_ND]     = loss;   // q_latent_loss
        out[VQ_ND + 1] = loss;   // e_latent_loss
    }
}

extern "C" void kernel_launch(void* const* d_in, const int* in_sizes, int n_in,
                              void* d_out, int out_size, void* d_ws, size_t ws_size,
                              hipStream_t stream) {
    const float* inputs   = (const float*)d_in[0];   // [8,2048,512] f32
    const float* codebook = (const float*)d_in[1];   // [8192,512]  f32
    float* out = (float*)d_out;

    char* ws = (char*)d_ws;
    _Float16* a16  = (_Float16*)(ws + WS_A16);
    _Float16* b16  = (_Float16*)(ws + WS_B16);
    float*    e2   = (float*)(ws + WS_E2);
    int*      idxw = (int*)(ws + WS_IDX);
    float*    acc  = (float*)(ws + WS_ACC);

    // 1) inputs -> f16, zero accumulator   (ND/8 threads)
    vq_convert_inputs<<<(int)(VQ_ND / 8 / 256), 256, 0, stream>>>(inputs, a16, acc);
    // 2) codebook -> f16 + norms           (one wave per code row)
    vq_convert_codebook<<<VQ_K / 8, 256, 0, stream>>>(codebook, b16, e2);
    // 3) WMMA distance + argmin            (128 rows / block, 128KB LDS)
    vq_argmin_wmma<<<VQ_N / 128, 256, 128 * VQ_D * (int)sizeof(_Float16), stream>>>(
        a16, b16, e2, idxw, out);
    // 4) gather + loss accumulation        (ND/4 threads)
    vq_gather_loss<<<(int)(VQ_ND / 4 / 256), 256, 0, stream>>>(inputs, codebook, idxw, out, acc);
    // 5) finalize losses
    vq_finalize<<<1, 32, 0, stream>>>(acc, out);
}
